// FeatureAttentionLayer_27212912787959
// MI455X (gfx1250) — matrix operand order repaired and checked by reading, not compiled, not run on hardware
//
#include <hip/hip_runtime.h>
#include <hip/hip_bf16.h>

typedef _Float16 v16h  __attribute__((ext_vector_type(16)));
typedef _Float16 half8 __attribute__((ext_vector_type(8)));
typedef float    v8f   __attribute__((ext_vector_type(8)));

#define ALPHA 0.2f
// b=8, w=256, K=256, E=512

// ---------------- prep kernels ----------------
__global__ void gat_prep_vh(const float* __restrict__ x, _Float16* __restrict__ vh) {
  int idx = blockIdx.x * blockDim.x + threadIdx.x;   // [0, 8*256*256)
  int b = idx >> 16;
  int k = (idx >> 8) & 255;
  int w = idx & 255;
  // vh[b][k][w] = x[b][w][k]  (the reference's v = transpose(x, (0,2,1)))
  vh[idx] = (_Float16)x[(b * 256 + w) * 256 + k];
}

__global__ void gat_prep_wh(const float* __restrict__ W, _Float16* __restrict__ Wh) {
  int idx = blockIdx.x * blockDim.x + threadIdx.x;   // [0, 512*512)
  Wh[idx] = (_Float16)W[idx];
}

// ---------------- WMMA GEMM: L = v*Wl^T, Rp = v*Wr^T + b_lin ----------------
// One wave per 16x16 output tile. 8192 tiles = 2 sides * 8 b * 16 itile * 32 etile.
__global__ void gat_gemm_wmma(const _Float16* __restrict__ vh,
                              const _Float16* __restrict__ Wh,
                              const float* __restrict__ b_lin,
                              float* __restrict__ Lbuf,
                              float* __restrict__ Rpbuf) {
  const int lane = threadIdx.x & 31;
  const int wave = threadIdx.x >> 5;
  const int tile = blockIdx.x * (blockDim.x >> 5) + wave;

  const int side  = tile >> 12;        // 0 = left half of W, 1 = right half
  const int rem   = tile & 4095;
  const int b     = rem >> 9;
  const int rem2  = rem & 511;
  const int itile = rem2 >> 5;         // 0..15  (K rows / 16)
  const int etile = rem2 & 31;         // 0..31  (E cols / 16)
  const int m0 = itile * 16;
  const int e0 = etile * 16;

  const int n  = lane & 15;            // A row index == B/C column index
  const int hi = lane >> 4;

  // A row: vh[b][m0+n][*]  (reduction dim w contiguous)
  const _Float16* arow = vh + ((b * 256 + m0 + n) * 256);
  // B row: Wh[e0+n][side*256 + *]
  const _Float16* brow0 = Wh + (e0 + n) * 512 + side * 256;

  v8f c = {};
#pragma unroll
  for (int w0 = 0; w0 < 256; w0 += 32) {
    // A (16x32 f16): lane holds contiguous K = hi*16 .. hi*16+15
    v16h a = *(const v16h*)(arow + w0 + hi * 16);
    // B (32x16 f16): lane n holds contiguous K = hi*16 .. hi*16+15 of column n
    v16h bm = *(const v16h*)(brow0 + w0 + hi * 16);
    c = __builtin_amdgcn_wmma_f32_16x16x32_f16(
        /*neg_a=*/false, a, /*neg_b=*/false, bm,
        /*c_mod=*/(short)0, c, /*reuse_a=*/false, /*reuse_b=*/false);
  }

  // C layout: VGPR r -> row m = hi*8 + r, column n
  if (side == 0) {
#pragma unroll
    for (int r = 0; r < 8; ++r) {
      int m = hi * 8 + r;
      Lbuf[((b * 256 + m0 + m) * 512) + e0 + n] = c[r];   // L[b][i][e]
    }
  } else {
    float bl = b_lin[e0 + n];
#pragma unroll
    for (int r = 0; r < 8; ++r) {
      int m = hi * 8 + r;
      // store transposed with b_lin folded in: Rp[b][e][j]
      Rpbuf[((b * 512 + e0 + n) * 256) + m0 + m] = c[r] + bl;
    }
  }
}

// ---------------- fused e-tensor + softmax + attn@v + sigmoid ----------------
// One 256-thread block per (b, i) row.
__global__ void gat_attn_fused(const float* __restrict__ Lbuf,
                               const float* __restrict__ Rpbuf,
                               const float* __restrict__ a_vec,
                               const float* __restrict__ bias,
                               const _Float16* __restrict__ vh,
                               float* __restrict__ out) {
  __shared__ float sL[512];
  __shared__ float sA[512];
  __shared__ float sRed[256];
  __shared__ float sAttn[256];

  const int tid = threadIdx.x;       // doubles as j index, then w index
  const int b = blockIdx.x >> 8;
  const int i = blockIdx.x & 255;

  const float* lrow = Lbuf + (b * 256 + i) * 512;
  sL[tid]       = lrow[tid];
  sL[tid + 256] = lrow[tid + 256];
  sA[tid]       = a_vec[tid];
  sA[tid + 256] = a_vec[tid + 256];
  __syncthreads();

  // e[b,i,j] = sum_e a_e * leakyrelu(L[i,e] + R[j,e] + b_lin[e])
  const float* rp = Rpbuf + b * 512 * 256 + tid;   // Rp[b][e][j=tid], coalesced
  float acc = 0.f;
#pragma unroll 4
  for (int e = 0; e < 512; ++e) {
    float t = sL[e] + rp[e * 256];
    float leaky = fmaxf(t, ALPHA * t);
    acc = fmaf(sA[e], leaky, acc);
  }
  float ej = acc + bias[i * 256 + tid];

  // softmax over j across the block
  sRed[tid] = ej;
  __syncthreads();
  for (int s = 128; s > 0; s >>= 1) {
    if (tid < s) sRed[tid] = fmaxf(sRed[tid], sRed[tid + s]);
    __syncthreads();
  }
  float mx = sRed[0];
  __syncthreads();
  float p = __expf(ej - mx);
  sRed[tid] = p;
  __syncthreads();
  for (int s = 128; s > 0; s >>= 1) {
    if (tid < s) sRed[tid] += sRed[tid + s];
    __syncthreads();
  }
  float inv = 1.0f / sRed[0];
  sAttn[tid] = p * inv;
  __syncthreads();

  // h[b][i][w] = sigmoid(sum_j attn[j] * v[b][j][w]),  w = tid
  const _Float16* vb = vh + b * 256 * 256 + tid;   // vh[b][j][w=tid], coalesced per j
  float acc2 = 0.f;
#pragma unroll 4
  for (int j = 0; j < 256; ++j) {
    acc2 = fmaf(sAttn[j], (float)vb[j * 256], acc2);
  }
  float h = 1.0f / (1.0f + __expf(-acc2));
  out[(b * 256 + tid) * 256 + i] = h;   // out[b][w][i] (final transpose)
}

// ---------------- launch ----------------
extern "C" void kernel_launch(void* const* d_in, const int* in_sizes, int n_in,
                              void* d_out, int out_size, void* d_ws, size_t ws_size,
                              hipStream_t stream) {
  const float* x     = (const float*)d_in[0];  // (8,256,256)
  const float* W     = (const float*)d_in[1];  // (512,512)
  const float* b_lin = (const float*)d_in[2];  // (512,)
  const float* a_vec = (const float*)d_in[3];  // (512,)
  const float* bias  = (const float*)d_in[4];  // (256,256)
  float* out = (float*)d_out;                  // (8,256,256)

  char* ws = (char*)d_ws;
  _Float16* vh   = (_Float16*)(ws);                       // 8*256*256 f16 = 1 MB
  _Float16* Wh   = (_Float16*)(ws + (1u << 20));          // 512*512 f16   = 0.5 MB
  float*    Lbuf = (float*)(ws + (1u << 20) + (1u << 19));        // 8*256*512 f32 = 4 MB
  float*    Rpbuf = Lbuf + 8 * 256 * 512;                         // 8*512*256 f32 = 4 MB

  gat_prep_vh<<<(8 * 256 * 256) / 256, 256, 0, stream>>>(x, vh);
  gat_prep_wh<<<(512 * 512) / 256, 256, 0, stream>>>(W, Wh);

  // 8192 tiles, 8 waves (256 threads) per block -> 1024 blocks
  gat_gemm_wmma<<<1024, 256, 0, stream>>>(vh, Wh, b_lin, Lbuf, Rpbuf);

  // one block per (b, i) row: 8*256 = 2048 blocks
  gat_attn_fused<<<2048, 256, 0, stream>>>(Lbuf, Rpbuf, a_vec, bias, vh, out);
}
